// Model1Net_84928683311203
// MI455X (gfx1250) — matrix operand order repaired and checked by reading, not compiled, not run on hardware
//
#include <hip/hip_runtime.h>

// ---------------------------------------------------------------------------
// MI455X (gfx1250) implementation.
// GEMMs      -> v_wmma_f32_16x16x32_bf16 (bf16 in, fp32 accum).
// Weights    -> staged into LDS once per workgroup by the Tensor Data Mover
//               (tensor_load_to_lds + s_wait_tensorcnt), with TDM row padding
//               (+16B/row) so B-fragment ds_load_b128 are bank-conflict free.
// Aggregation-> global_atomic_add_f32 (the memory-bound part of the model).
// wave32 everywhere; reductions via __shfl_xor over 32 lanes.
// ---------------------------------------------------------------------------

typedef __attribute__((ext_vector_type(16))) __bf16       v16bf;
typedef __attribute__((ext_vector_type(8)))  float        v8f;
typedef __attribute__((ext_vector_type(4)))  unsigned int v4u;
typedef __attribute__((ext_vector_type(8)))  int          v8i;
typedef __attribute__((ext_vector_type(4)))  int          v4i;

#define HID 128

enum { GF_RELU = 1, GF_BIAS = 2, GF_ADD = 4, GF_WF32 = 8, GF_WBF = 16 };

// C[M x 128] = act( A[M x K](bf16, row stride lda) @ W[128 x K]^T(bf16) + bias + addend )
// One wave computes a 16x128 strip: 8 WMMA accumulators. Block = 256 thr = 8 waves.
// K must be a multiple of 32. W is staged to LDS by the TDM in 128-column blocks,
// each row padded by 16B (pad_interval/pad_amount in the D#).
//
// A fragment (16-bit A 16x32): lane = 16*kh + ml holds row ml,
//   elems 0..15 = K[k0+16*kh .. +15] (contiguous per lane-half, matches B).
// B fragment (32x16): lane = 16*kh + nl holds column nl,
//   elems 0..15 = K[k0+16*kh .. +15] (contiguous per lane-half).
__global__ __launch_bounds__(256)
void k_gemm128(const __bf16* __restrict__ A, int lda, int M, int K,
               const __bf16* __restrict__ W, const float* __restrict__ bias,
               const float* __restrict__ addend,
               float* __restrict__ Cf, __bf16* __restrict__ Cbf, int flags)
{
    extern __shared__ __align__(16) char smem[];
    const int tile0     = (K < 128) ? K : 128;   // columns per TDM block
    const int rowStride = tile0 * 2 + 16;        // bytes; +16B kills bank conflicts
    const int blkBytes  = 128 * rowStride;
    const int nblk      = K / tile0;

    // ---- TDM: DMA W (128 x K bf16) into LDS; one D# per 128-column block ----
    if (threadIdx.x < 32) {
        const unsigned ldsBase = (unsigned)(size_t)smem;       // low 32b = LDS offset
        const unsigned pi = (tile0 == 128) ? 5u : 3u;          // 64 / 16 dwords per row
        for (int c = 0; c < nblk; ++c) {
            unsigned long long ga = (unsigned long long)(const void*)(W + (size_t)c * tile0);
            v4u g0;
            g0[0] = 1u;                                        // count=1, user mode
            g0[1] = ldsBase + (unsigned)(c * blkBytes);        // lds_addr
            g0[2] = (unsigned)(ga & 0xFFFFFFFFu);              // global_addr[31:0]
            g0[3] = (unsigned)((ga >> 32) & 0x01FFFFFFu) | (2u << 30); // [56:32] | type=2
            v8i g1;
            g1[0] = (int)((1u << 16) | (1u << 20) | (pi << 22) | (3u << 25));
            //        data_size=2B | pad_enable | pad_interval | pad_amount=4 dwords (16B)
            g1[1] = (int)(((unsigned)K & 0xFFFFu) << 16);              // tensor_dim0 lo16
            g1[2] = (int)((((unsigned)K >> 16) & 0xFFFFu) | (128u << 16)); // dim0 hi | dim1 lo
            g1[3] = (int)(((unsigned)tile0 & 0xFFFFu) << 16);          // dim1 hi=0 | tile_dim0
            g1[4] = 128;                                               // tile_dim1 | tile_dim2=0
            g1[5] = K;                                                 // tensor_dim0_stride lo32
            g1[6] = 0;                                                 // stride hi | dim1_stride lo
            g1[7] = 0;
            v4i z4 = {};
#if __clang_major__ >= 23
            v8i z8 = {};
            __builtin_amdgcn_tensor_load_to_lds(g0, g1, z4, z4, z8, 0);
#else
            __builtin_amdgcn_tensor_load_to_lds(g0, g1, z4, z4, 0);
#endif
        }
        __builtin_amdgcn_s_wait_tensorcnt(0);
    }
    __syncthreads();   // all threads reach this (no early return above)

    const int lane = threadIdx.x & 31;
    const int wv   = threadIdx.x >> 5;
    const int m0   = (blockIdx.x * 8 + wv) * 16;
    if (m0 >= M) return;
    const int ml = lane & 15;
    const int kh = lane >> 4;

    const __bf16* arow = A + (size_t)(m0 + ml) * lda;

    v8f acc[8] = {};
    union Frag { uint4 u[2]; v16bf v; };

    for (int k0 = 0; k0 < K; k0 += 32) {
        const int  c    = k0 >> 7;          // 128-col block index (0 when K<=128)
        const int  kc   = k0 & 127;         // column within block
        const char* base = smem + c * blkBytes + (kc + 16 * kh) * 2;
        Frag a;
        a.u[0] = *(const uint4*)(arow + k0 + 16 * kh);
        a.u[1] = *(const uint4*)(arow + k0 + 16 * kh + 8);
        Frag b[8];
#pragma unroll
        for (int t = 0; t < 8; ++t) {       // preload all B frags -> one wait, 8 WMMAs
            const char* wrow = base + (t * 16 + ml) * rowStride;
            b[t].u[0] = *(const uint4*)(wrow);
            b[t].u[1] = *(const uint4*)(wrow + 16);
        }
#pragma unroll
        for (int t = 0; t < 8; ++t)
            acc[t] = __builtin_amdgcn_wmma_f32_16x16x32_bf16(
                false, a.v, false, b[t].v, (short)0, acc[t], false, false);
    }

    // Epilogue. C/D layout: lane 16*kh+ml, VGPR r -> (m = m0 + r + 8*kh, n = t*16 + ml)
#pragma unroll
    for (int t = 0; t < 8; ++t) {
        const int n = t * 16 + ml;
        const float bv = (flags & GF_BIAS) ? bias[n] : 0.0f;
#pragma unroll
        for (int r = 0; r < 8; ++r) {
            const int m = m0 + r + 8 * kh;
            float x = acc[t][r] + bv;
            if (flags & GF_ADD)  x += addend[(size_t)m * HID + n];
            if (flags & GF_RELU) x = x > 0.0f ? x : 0.0f;
            if (flags & GF_WF32) Cf[(size_t)m * HID + n] = x;
            if (flags & GF_WBF)  Cbf[(size_t)m * HID + n] = (__bf16)x;
        }
    }
}

// fp32 -> bf16 cast with optional K padding to ldd (zeros past K).
__global__ void k_cast_pad(const float* __restrict__ src, __bf16* __restrict__ dst,
                           int M, int K, int ldd)
{
    size_t i = (size_t)blockIdx.x * blockDim.x + threadIdx.x;
    if (i >= (size_t)M * ldd) return;
    int k = (int)(i % ldd);
    size_t m = i / ldd;
    dst[i] = (k < K) ? (__bf16)src[m * K + k] : (__bf16)0.0f;
}

// Segment-sum scatter: one wave per edge; lane handles 4 features via float4
// gather + 4 global fp32 atomics. deg accumulated once per edge.
__global__ __launch_bounds__(256)
void k_scatter_add(const float* __restrict__ h, const int* __restrict__ ei, int E,
                   float* __restrict__ agg, float* __restrict__ deg)
{
    int e = blockIdx.x * 8 + (threadIdx.x >> 5);
    if (e >= E) return;
    int lane = threadIdx.x & 31;
    int src = ei[e];
    int dst = ei[E + e];
    float4 v = *(const float4*)(h + (size_t)src * HID + lane * 4);
    float* ad = agg + (size_t)dst * HID + lane * 4;
    atomicAdd(ad + 0, v.x); atomicAdd(ad + 1, v.y);
    atomicAdd(ad + 2, v.z); atomicAdd(ad + 3, v.w);
    if (lane == 0) atomicAdd(deg + dst, 1.0f);
}

// mean = agg / max(deg,1), emitted as bf16 (only consumed as a WMMA A operand).
__global__ void k_mean_cast(const float* __restrict__ agg, const float* __restrict__ deg,
                            __bf16* __restrict__ outb, int M)
{
    size_t i = (size_t)blockIdx.x * blockDim.x + threadIdx.x;
    if (i >= (size_t)M * HID) return;
    float d = deg[i >> 7];
    d = d > 1.0f ? d : 1.0f;
    outb[i] = (__bf16)(agg[i] / d);
}

// h = LayerNorm(relu(pre) + h_in)*g + b ; writes fp32 master + bf16 copy.
// One wave (32 lanes) per row of 128; shuffle reductions.
__global__ __launch_bounds__(128)
void k_relu_add_ln(const float* __restrict__ pre, const float* __restrict__ hin,
                   const float* __restrict__ g, const float* __restrict__ b,
                   float* __restrict__ hout, __bf16* __restrict__ hbf, int M)
{
    int row = blockIdx.x * 4 + (threadIdx.x >> 5);
    if (row >= M) return;
    int lane = threadIdx.x & 31;
    const float* p  = pre + (size_t)row * HID;
    const float* hi = hin + (size_t)row * HID;
    float x[4];
    float s = 0.0f;
#pragma unroll
    for (int i = 0; i < 4; ++i) {
        float v = p[lane * 4 + i];
        v = v > 0.0f ? v : 0.0f;
        v += hi[lane * 4 + i];
        x[i] = v; s += v;
    }
    for (int off = 16; off > 0; off >>= 1) s += __shfl_xor(s, off, 32);
    float mean = s * (1.0f / 128.0f);
    float vs = 0.0f;
#pragma unroll
    for (int i = 0; i < 4; ++i) { float d = x[i] - mean; vs += d * d; }
    for (int off = 16; off > 0; off >>= 1) vs += __shfl_xor(vs, off, 32);
    float rstd = rsqrtf(vs * (1.0f / 128.0f) + 1e-5f);
#pragma unroll
    for (int i = 0; i < 4; ++i) {
        int j = lane * 4 + i;
        float y = (x[i] - mean) * rstd * g[j] + b[j];
        hout[(size_t)row * HID + j] = y;
        hbf[(size_t)row * HID + j]  = (__bf16)y;
    }
}

// out[m] = dot(h[m,:], w3) + b3  (final 1x128 head layer). One wave per row.
__global__ __launch_bounds__(128)
void k_final_dot(const float* __restrict__ h, const float* __restrict__ w3,
                 const float* __restrict__ b3, float* __restrict__ out, int M)
{
    int row = blockIdx.x * 4 + (threadIdx.x >> 5);
    if (row >= M) return;
    int lane = threadIdx.x & 31;
    float4 hv = *(const float4*)(h + (size_t)row * HID + lane * 4);
    float4 wv = *(const float4*)(w3 + lane * 4);
    float s = hv.x * wv.x + hv.y * wv.y + hv.z * wv.z + hv.w * wv.w;
    for (int off = 16; off > 0; off >>= 1) s += __shfl_xor(s, off, 32);
    if (lane == 0) out[row] = s + b3[0];
}

// Build [E, 384] bf16 = [h1[src] | h1[dst] | h_e] for head_edge GEMM (K=384).
__global__ void k_edge_concat(const __bf16* __restrict__ h1b, const __bf16* __restrict__ heb,
                              const int* __restrict__ ei, int E, __bf16* __restrict__ out)
{
    size_t i = (size_t)blockIdx.x * blockDim.x + threadIdx.x;
    if (i >= (size_t)E * 384) return;
    int j = (int)(i % 384);
    size_t e = i / 384;
    __bf16 v;
    if (j < 128)      v = h1b[(size_t)ei[e] * HID + j];
    else if (j < 256) v = h1b[(size_t)ei[E + e] * HID + (j - 128)];
    else              v = heb[e * HID + (j - 256)];
    out[i] = v;
}

// ---------------------------------------------------------------------------
static inline int cdiv(long long a, long long b) { return (int)((a + b - 1) / b); }

extern "C" void kernel_launch(void* const* d_in, const int* in_sizes, int n_in,
                              void* d_out, int out_size, void* d_ws, size_t ws_size,
                              hipStream_t stream)
{
    (void)n_in; (void)out_size;
    const int N2 = in_sizes[0] / 16;   // x2d [N2,16]
    const int E2 = in_sizes[1] / 2;    // edge_index_2d [2,E2]
    const int N1 = in_sizes[2] / 12;   // x1d [N1,12]
    const int E1 = in_sizes[3] / 2;    // edge_index_1d [2,E1]

    const float* x2d = (const float*)d_in[0];
    const int*   ei2 = (const int*)d_in[1];
    const float* x1d = (const float*)d_in[2];
    const int*   ei1 = (const int*)d_in[3];
    const float* xe  = (const float*)d_in[4];
    // params flattened in dict insertion order:
    // 5: gnn2d.in_w, 6: in_b, 7+5l: {wl,bl,wr,g,b} l=0..2
    // 22..27: head2d {w1,b1,w2,b2,w3,b3}
    // 28: gnn1d.in_w, 29: in_b, 30+5l: layers l=0..1
    // 40..45: head1d, 46..51: head_inlet, 52: edge_w, 53: edge_b, 54..59: head_edge
    auto P = [&](int i) { return (const float*)d_in[i]; };

    // ---- workspace carve-out ----
    char* wsp = (char*)d_ws;
    auto alloc = [&](size_t bytes) -> void* {
        void* p = (void*)wsp;
        wsp += (bytes + 255) & ~(size_t)255;
        return p;
    };
    float*  h2f  = (float*)alloc((size_t)N2 * HID * 4);
    float*  tmpC = (float*)alloc((size_t)N2 * HID * 4);   // agg / GEMM out (also E1x128)
    float*  tmpD = (float*)alloc((size_t)N2 * HID * 4);
    __bf16* h2b  = (__bf16*)alloc((size_t)N2 * HID * 2);
    __bf16* sb1  = (__bf16*)alloc((size_t)N2 * HID * 2);  // bf16 scratch (mean / head t1)
    __bf16* sb2  = (__bf16*)alloc((size_t)N2 * HID * 2);
    float*  deg2 = (float*)alloc((size_t)N2 * 4);
    float*  h1f  = (float*)alloc((size_t)N1 * HID * 4);
    __bf16* h1b  = (__bf16*)alloc((size_t)N1 * HID * 2);
    float*  deg1 = (float*)alloc((size_t)N1 * 4);
    __bf16* x2p  = (__bf16*)alloc((size_t)N2 * 32 * 2);   // inputs padded to K=32
    __bf16* x1p  = (__bf16*)alloc((size_t)N1 * 32 * 2);
    __bf16* xep  = (__bf16*)alloc((size_t)E1 * 32 * 2);
    __bf16* cat  = (__bf16*)alloc((size_t)E1 * 384 * 2);  // [h1[src]|h1[dst]|h_e]
    __bf16* wreg = (__bf16*)alloc((size_t)1 << 20);       // bf16 weights (~680 KB)
    if ((size_t)(wsp - (char*)d_ws) > ws_size) return;    // not enough scratch

    // ---- helper launchers ----
    auto cast = [&](const float* s, __bf16* d, int M, int K, int ld) {
        long long n = (long long)M * ld;
        hipLaunchKernelGGL(k_cast_pad, dim3(cdiv(n, 256)), dim3(256), 0, stream, s, d, M, K, ld);
    };
    __bf16* wcur = wreg;
    auto castw = [&](int idx, int M, int K, int ld) -> const __bf16* {
        __bf16* r = wcur;
        cast(P(idx), r, M, K, ld);
        wcur += (size_t)M * ld;
        return r;
    };
    auto gemm = [&](const __bf16* A, int lda, int M, int K, const __bf16* W,
                    const float* bias, const float* add, float* Cf, __bf16* Cbf, int flags) {
        int tile0 = (K < 128) ? K : 128;
        size_t shmem = (size_t)(K / tile0) * 128 * (tile0 * 2 + 16);
        hipLaunchKernelGGL(k_gemm128, dim3(cdiv(M, 128)), dim3(256), shmem, stream,
                           A, lda, M, K, W, bias, add, Cf, Cbf, flags);
    };

    // ---- one-time-per-call conversions (tiny; weights live in L2 thereafter) ----
    cast(x2d, x2p, N2, 16, 32);
    cast(x1d, x1p, N1, 12, 32);
    cast(xe,  xep, E1, 10, 32);
    const __bf16* w_in2 = castw(5, HID, 16, 32);
    const __bf16 *w2_l[3], *w2_r[3];
    for (int l = 0; l < 3; ++l) { w2_l[l] = castw(7 + 5 * l, HID, HID, HID);
                                  w2_r[l] = castw(9 + 5 * l, HID, HID, HID); }
    const __bf16* h2w1 = castw(22, HID, HID, HID);
    const __bf16* h2w2 = castw(24, HID, HID, HID);
    const __bf16* w_in1 = castw(28, HID, 12, 32);
    const __bf16 *w1_l[2], *w1_r[2];
    for (int l = 0; l < 2; ++l) { w1_l[l] = castw(30 + 5 * l, HID, HID, HID);
                                  w1_r[l] = castw(32 + 5 * l, HID, HID, HID); }
    const __bf16* h1w1 = castw(40, HID, HID, HID);
    const __bf16* h1w2 = castw(42, HID, HID, HID);
    const __bf16* hiw1 = castw(46, HID, HID, HID);
    const __bf16* hiw2 = castw(48, HID, HID, HID);
    const __bf16* wew  = castw(52, HID, 10, 32);
    const __bf16* hew1 = castw(54, HID, 384, 384);
    const __bf16* hew2 = castw(56, HID, HID, HID);

    // ---- SAGE stack runner ----
    auto sage = [&](int M, int E, const int* ei, const __bf16* xp, const __bf16* win,
                    const float* bin, int lbase, int nlayers, const __bf16* const* wl,
                    const __bf16* const* wr, float* hf, __bf16* hb, float* deg) {
        // h = x @ in_w.T + in_b
        gemm(xp, 32, M, 32, win, bin, nullptr, hf, hb, GF_BIAS | GF_WF32 | GF_WBF);
        for (int l = 0; l < nlayers; ++l) {
            const float* bl = P(lbase + 5 * l + 1);
            const float* g  = P(lbase + 5 * l + 3);
            const float* b  = P(lbase + 5 * l + 4);
            hipMemsetAsync(tmpC, 0, (size_t)M * HID * 4, stream);
            hipMemsetAsync(deg,  0, (size_t)M * 4, stream);
            hipLaunchKernelGGL(k_scatter_add, dim3(cdiv(E, 8)), dim3(256), 0, stream,
                               hf, ei, E, tmpC, deg);
            hipLaunchKernelGGL(k_mean_cast, dim3(cdiv((long long)M * HID, 256)), dim3(256),
                               0, stream, tmpC, deg, sb1, M);
            gemm(hb, HID, M, HID, wr[l], nullptr, nullptr, tmpD, nullptr, GF_WF32);
            gemm(sb1, HID, M, HID, wl[l], bl, tmpD, tmpC, nullptr, GF_BIAS | GF_ADD | GF_WF32);
            hipLaunchKernelGGL(k_relu_add_ln, dim3(cdiv(M, 4)), dim3(128), 0, stream,
                               tmpC, hf, g, b, hf, hb, M);
        }
    };
    // ---- MLP head: out = (relu(relu(h@w1.T+b1)@w2.T+b2))@w3.T+b3 ----
    auto head = [&](int M, const __bf16* hb, const __bf16* w1b, const float* b1,
                    const __bf16* w2b, const float* b2, const float* w3, const float* b3,
                    float* out, int K1) {
        gemm(hb, K1, M, K1, w1b, b1, nullptr, nullptr, sb1, GF_BIAS | GF_RELU | GF_WBF);
        gemm(sb1, HID, M, HID, w2b, b2, nullptr, tmpD, nullptr, GF_BIAS | GF_RELU | GF_WF32);
        hipLaunchKernelGGL(k_final_dot, dim3(cdiv(M, 4)), dim3(128), 0, stream,
                           tmpD, w3, b3, out, M);
    };

    float* out = (float*)d_out;

    // === 2D branch ===
    sage(N2, E2, ei2, x2p, w_in2, P(6), 7, 3, w2_l, w2_r, h2f, h2b, deg2);
    head(N2, h2b, h2w1, P(23), h2w2, P(25), P(26), P(27), out, HID);               // d2

    // === 1D branch ===
    sage(N1, E1, ei1, x1p, w_in1, P(29), 30, 2, w1_l, w1_r, h1f, h1b, deg1);
    head(N1, h1b, h1w1, P(41), h1w2, P(43), P(44), P(45), out + N2, HID);          // d1
    head(N1, h1b, hiw1, P(47), hiw2, P(49), P(50), P(51), out + N2 + N1, HID);     // inlet

    // === edge head ===
    // h_e = relu(x_edge @ edge_w.T + edge_b)  -> bf16 into sb2 (sb1 is head scratch)
    gemm(xep, 32, E1, 32, wew, P(53), nullptr, nullptr, sb2, GF_BIAS | GF_RELU | GF_WBF);
    hipLaunchKernelGGL(k_edge_concat, dim3(cdiv((long long)E1 * 384, 256)), dim3(256),
                       0, stream, h1b, sb2, ei1, E1, cat);
    head(E1, cat, hew1, P(55), hew2, P(57), P(58), P(59), out + N2 + 2 * N1, 384); // eflow
}
